// Architecture_30554397344119
// MI455X (gfx1250) — matrix operand order repaired and checked by reading, not compiled, not run on hardware
//
#include <hip/hip_runtime.h>
#include <hip/hip_bf16.h>

typedef __attribute__((ext_vector_type(16))) _Float16 v16h;
typedef __attribute__((ext_vector_type(8)))  float    v8f;
typedef __attribute__((ext_vector_type(4)))  int      v4i;

// ---------------------------------------------------------------------------
// Model constants
// ---------------------------------------------------------------------------
#define AKT_D   512
#define AKT_H   8
#define AKT_DH  64
#define AKT_F   2048
#define AKT_B   32
#define AKT_S   512
#define AKT_BS  (AKT_B * AKT_S)

// ---------------------------------------------------------------------------
// CDNA5 async HBM->LDS copy (ASYNCcnt path), with compile-safe fallbacks.
// Builtin signature (from hipcc diagnostic): (int4 AS1*, int4 AS3*, Iimm, Iimm)
// ---------------------------------------------------------------------------
#if __has_builtin(__builtin_amdgcn_global_load_async_to_lds_b128)
#define HAVE_ASYNC_LDS 1
#else
#define HAVE_ASYNC_LDS 0
#endif

__device__ __forceinline__ void async_copy16(const float* g, float* l) {
#if HAVE_ASYNC_LDS
  float* gnc = const_cast<float*>(g);
  __builtin_amdgcn_global_load_async_to_lds_b128(
      (__attribute__((address_space(1))) v4i*)gnc,
      (__attribute__((address_space(3))) v4i*)l, 0, 0);
#else
  *(float4*)l = *(const float4*)g;
#endif
}

#if __has_builtin(__builtin_amdgcn_s_wait_asynccnt)
#define WAIT_ASYNC(n) __builtin_amdgcn_s_wait_asynccnt(n)
#elif HAVE_ASYNC_LDS
#define WAIT_ASYNC(n) asm volatile("s_wait_asynccnt %0" ::"i"(n) : "memory")
#else
#define WAIT_ASYNC(n) ((void)0)
#endif

// ---------------------------------------------------------------------------
// GEMM: C[M,N] = A[M,K] (row-major, fp32) x W[N,K]^T (row-major, fp32) + bias
// f16 WMMA 16x16x32, fp32 accumulate.  Tile 128x64x64, 8 waves, each wave
// 32x32 output (2x2 WMMA), double-buffered LDS fed by async DMA.
// ---------------------------------------------------------------------------
#define TM 128
#define TN 64
#define TK 64
#define LPAD 4  // +4 floats -> 68-dword row stride (conflict-free, 16B aligned)

__global__ __launch_bounds__(256) void gemm_wmma_kernel(
    const float* __restrict__ A, const float* __restrict__ W,
    const float* __restrict__ bias, float* __restrict__ C,
    int M, int N, int K, int relu)
{
  __shared__ float As[2][TM][TK + LPAD];
  __shared__ float Bs[2][TN][TK + LPAD];

  const int tid  = threadIdx.x;
  const int lane = tid & 31;
  const int wave = tid >> 5;
  const int m0 = blockIdx.y * TM;
  const int n0 = blockIdx.x * TN;
  const int wm = (wave >> 1) * 32;   // 4 wave-rows
  const int wn = (wave & 1) * 32;    // 2 wave-cols

  const int mA = lane & 15;
  const int hA = lane >> 4;

  // Issue one tile's worth of 16-byte async segments (12 per thread/wave).
  auto issue_tile = [&](int kt, int buf) {
#pragma unroll
    for (int i = 0; i < 8; ++i) {                 // A: 128x64 fp32 = 2048 segs
      int seg = tid + i * 256;
      int r = seg >> 4, sc = (seg & 15) * 4;
      async_copy16(&A[(size_t)(m0 + r) * K + kt + sc], &As[buf][r][sc]);
    }
#pragma unroll
    for (int i = 0; i < 4; ++i) {                 // B: 64x64 fp32 = 1024 segs
      int seg = tid + i * 256;
      int r = seg >> 4, sc = (seg & 15) * 4;
      async_copy16(&W[(size_t)(n0 + r) * K + kt + sc], &Bs[buf][r][sc]);
    }
  };

  v8f acc[2][2] = {};
  const int nk = K / TK;

  issue_tile(0, 0);
  for (int kt = 0; kt < nk; ++kt) {
    const int buf = kt & 1;
    if (kt + 1 < nk) {
      issue_tile((kt + 1) * TK, buf ^ 1);   // prefetch next tile
      WAIT_ASYNC(12);                       // wait: current tile's 12 issues done
    } else {
      WAIT_ASYNC(0);
    }
    __syncthreads();

#pragma unroll
    for (int kc = 0; kc < 2; ++kc) {
      v16h af[2], bf[2];
#pragma unroll
      for (int t = 0; t < 2; ++t) {
#pragma unroll
        for (int j = 0; j < 8; ++j) {
          // A fragment: lanes 0-15 K=0..7,16..23 ; lanes 16-31 K=8..15,24..31
          int ka = kc * 32 + (j & 3) * 2 + 8 * hA + ((j >> 2) << 4);
          af[t][2 * j]     = (_Float16)As[buf][wm + t * 16 + mA][ka];
          af[t][2 * j + 1] = (_Float16)As[buf][wm + t * 16 + mA][ka + 1];
          // B fragment: lanes 0-15 K=0..7,16..23 ; lanes 16-31 K=8..15,24..31
          int kb = kc * 32 + (j & 3) * 2 + 8 * hA + ((j >> 2) << 4);
          bf[t][2 * j]     = (_Float16)Bs[buf][wn + t * 16 + mA][kb];
          bf[t][2 * j + 1] = (_Float16)Bs[buf][wn + t * 16 + mA][kb + 1];
        }
      }
#pragma unroll
      for (int i = 0; i < 2; ++i)
#pragma unroll
        for (int j = 0; j < 2; ++j)
          acc[i][j] = __builtin_amdgcn_wmma_f32_16x16x32_f16(
              false, af[i], false, bf[j], (short)0, acc[i][j], false, false);
    }
    __syncthreads();   // all waves done with buf before its async overwrite
  }

  // Epilogue: C layout -> VGPR r holds M = r + 8*h, N = lane&15
#pragma unroll
  for (int i = 0; i < 2; ++i)
#pragma unroll
    for (int j = 0; j < 2; ++j) {
      int col = n0 + wn + j * 16 + mA;
      float bv = bias[col];
#pragma unroll
      for (int r = 0; r < 8; ++r) {
        int row = m0 + wm + i * 16 + r + 8 * hA;
        float v = acc[i][j][r] + bv;
        if (relu) v = fmaxf(v, 0.f);
        C[(size_t)row * N + col] = v;
      }
    }
}

// ---------------------------------------------------------------------------
// Fused AKT attention: one wave handles one (b, h, 16-query-row) block.
// scores via WMMA -> masked softmax -> cumsum distance decay -> softmax ->
// attn x V via WMMA.  Q/K/V/O are [B,S,D] with head h at columns h*64..h*64+63.
// ---------------------------------------------------------------------------
__global__ __launch_bounds__(32) void akt_attn_kernel(
    const float* __restrict__ Q, const float* __restrict__ K,
    const float* __restrict__ V, const float* __restrict__ gammas,
    float* __restrict__ O, int mask_flag)
{
  __shared__ float Srow[16][AKT_S + 4];

  const int lane = threadIdx.x & 31;
  const int bh = blockIdx.y;
  const int b = bh / AKT_H, h = bh % AKT_H;
  const int q0 = blockIdx.x * 16;
  const size_t base = (size_t)b * AKT_S * AKT_D + (size_t)h * AKT_DH;

  const int mA = lane & 15;
  const int hA = lane >> 4;

  // Q fragments (16x64 in two 16x32 chunks), fp32 -> f16
  v16h qf[2];
#pragma unroll
  for (int c = 0; c < 2; ++c) {
    const float* qr = Q + base + (size_t)(q0 + mA) * AKT_D;
#pragma unroll
    for (int j = 0; j < 8; ++j) {
      int ka = (j & 3) * 2 + 8 * hA + ((j >> 2) << 4) + 32 * c;
      qf[c][2 * j]     = (_Float16)qr[ka];
      qf[c][2 * j + 1] = (_Float16)qr[ka + 1];
    }
  }

  const int qmaxrow = q0 + 15;
  const int kallow  = mask_flag ? qmaxrow : (qmaxrow - 1);   // >= 14 always
  const int ntile   = kallow / 16 + 1;
  const int nchunk  = kallow / 32 + 1;

  // --- scores = Q K^T / 8 into LDS -------------------------------------
  for (int kt = 0; kt < ntile; ++kt) {
    v16h kf[2];
#pragma unroll
    for (int c = 0; c < 2; ++c) {
      const float* kr = K + base + (size_t)(kt * 16 + mA) * AKT_D;
#pragma unroll
      for (int j = 0; j < 8; ++j) {
        int kd = (j & 3) * 2 + 8 * hA + ((j >> 2) << 4) + 32 * c;
        kf[c][2 * j]     = (_Float16)kr[kd];
        kf[c][2 * j + 1] = (_Float16)kr[kd + 1];
      }
    }
    v8f sc = {};
    sc = __builtin_amdgcn_wmma_f32_16x16x32_f16(false, qf[0], false, kf[0],
                                                (short)0, sc, false, false);
    sc = __builtin_amdgcn_wmma_f32_16x16x32_f16(false, qf[1], false, kf[1],
                                                (short)0, sc, false, false);
#pragma unroll
    for (int r = 0; r < 8; ++r)
      Srow[r + 8 * hA][kt * 16 + mA] = sc[r] * 0.125f;   // 1/sqrt(64)
  }

  // --- per-row softmax + distance decay + softmax ----------------------
  const float gamma = -log1pf(__expf(gammas[h]));        // -softplus
  for (int m = 0; m < 16; ++m) {
    const int q = q0 + m;
    const int kmax = mask_flag ? q : (q - 1);
    float v[16], p[16];
#pragma unroll
    for (int i = 0; i < 16; ++i) v[i] = Srow[m][lane * 16 + i];

    float mx = -3.0e38f;
#pragma unroll
    for (int i = 0; i < 16; ++i)
      if (lane * 16 + i <= kmax) mx = fmaxf(mx, v[i]);
    for (int off = 16; off > 0; off >>= 1) mx = fmaxf(mx, __shfl_xor(mx, off));

    float sum = 0.f;
#pragma unroll
    for (int i = 0; i < 16; ++i) {
      p[i] = (lane * 16 + i <= kmax) ? __expf(v[i] - mx) : 0.f;
      sum += p[i];
    }
    float tot = sum;
    for (int off = 16; off > 0; off >>= 1) tot += __shfl_xor(tot, off);
    const float inv = 1.0f / tot;

    // cumulative sum of normalized scores (lane-local + cross-lane scan)
    float cs[16];
    float run = 0.f;
#pragma unroll
    for (int i = 0; i < 16; ++i) { run += p[i] * inv; cs[i] = run; }
    float inc = run;
    for (int off = 1; off < 32; off <<= 1) {
      float y = __shfl_up(inc, off);
      if (lane >= off) inc += y;
    }
    const float excl = inc - run;
    const float disttot = __shfl(inc, 31);

    float w2[16];
    float mx2 = -3.0e38f;
#pragma unroll
    for (int i = 0; i < 16; ++i) {
      int col = lane * 16 + i;
      float distcum = excl + cs[i];
      float pos = fabsf((float)(q - col));
      float dd = (disttot - distcum) * pos;
      dd = dd > 0.f ? dd : 0.f;
      float te = __expf(sqrtf(dd) * gamma);
      te = fminf(fmaxf(te, 1e-5f), 1e5f);
      w2[i] = (col <= kmax) ? v[i] * te : -3.0e38f;
      mx2 = fmaxf(mx2, w2[i]);
    }
    for (int off = 16; off > 0; off >>= 1) mx2 = fmaxf(mx2, __shfl_xor(mx2, off));
    float s2 = 0.f;
#pragma unroll
    for (int i = 0; i < 16; ++i) {
      w2[i] = (lane * 16 + i <= kmax) ? __expf(w2[i] - mx2) : 0.f;
      s2 += w2[i];
    }
    for (int off = 16; off > 0; off >>= 1) s2 += __shfl_xor(s2, off);
    const float inv2 = 1.0f / s2;
    const bool zero_row = (!mask_flag) && (q == 0);       // zero_pad
#pragma unroll
    for (int i = 0; i < 16; ++i)
      Srow[m][lane * 16 + i] = zero_row ? 0.f : w2[i] * inv2;
  }

  // --- out = attn x V ---------------------------------------------------
  v8f o[4] = {v8f{}, v8f{}, v8f{}, v8f{}};
  for (int kc = 0; kc < nchunk; ++kc) {
    v16h af;
#pragma unroll
    for (int j = 0; j < 8; ++j) {
      int ka = (j & 3) * 2 + 8 * hA + ((j >> 2) << 4);
      af[2 * j]     = (_Float16)Srow[mA][kc * 32 + ka];
      af[2 * j + 1] = (_Float16)Srow[mA][kc * 32 + ka + 1];
    }
#pragma unroll
    for (int t = 0; t < 4; ++t) {
      v16h bf;
#pragma unroll
      for (int j = 0; j < 8; ++j) {
        int kd = kc * 32 + (j & 3) * 2 + 8 * hA + ((j >> 2) << 4);
        bf[2 * j]     = (_Float16)V[base + (size_t)kd * AKT_D + t * 16 + mA];
        bf[2 * j + 1] = (_Float16)V[base + (size_t)(kd + 1) * AKT_D + t * 16 + mA];
      }
      o[t] = __builtin_amdgcn_wmma_f32_16x16x32_f16(false, af, false, bf,
                                                    (short)0, o[t], false, false);
    }
  }
#pragma unroll
  for (int t = 0; t < 4; ++t)
#pragma unroll
    for (int r = 0; r < 8; ++r)
      O[base + (size_t)(q0 + r + 8 * hA) * AKT_D + t * 16 + mA] = o[t][r];
}

// ---------------------------------------------------------------------------
// out[row] = LayerNorm(X[row] + R[row]) * g + b   (one wave per 512-wide row)
// ---------------------------------------------------------------------------
__global__ __launch_bounds__(32) void residual_ln_kernel(
    const float* __restrict__ X, const float* __restrict__ R,
    const float* __restrict__ g, const float* __restrict__ bta,
    float* __restrict__ out)
{
  const int row  = blockIdx.x;
  const int lane = threadIdx.x;
  const size_t rb = (size_t)row * AKT_D;

  float x[16];
  float s = 0.f;
#pragma unroll
  for (int i = 0; i < 16; ++i) {
    x[i] = X[rb + lane * 16 + i] + R[rb + lane * 16 + i];
    s += x[i];
  }
  for (int off = 16; off > 0; off >>= 1) s += __shfl_xor(s, off);
  const float mu = s * (1.f / AKT_D);
  float vs = 0.f;
#pragma unroll
  for (int i = 0; i < 16; ++i) { float d = x[i] - mu; vs += d * d; }
  for (int off = 16; off > 0; off >>= 1) vs += __shfl_xor(vs, off);
  const float rinv = rsqrtf(vs * (1.f / AKT_D) + 1e-5f);
#pragma unroll
  for (int i = 0; i < 16; ++i) {
    int c = lane * 16 + i;
    out[rb + c] = (x[i] - mu) * rinv * g[c] + bta[c];
  }
}

// ---------------------------------------------------------------------------
// Host orchestration
// ---------------------------------------------------------------------------
struct LayerP {
  const float *Wk, *bk, *Wv, *bv, *Wo, *bo, *gammas;
  const float *ln1g, *ln1b, *W1, *b1, *W2, *b2, *ln2g, *ln2b;
};

static LayerP parse_layer(void* const* d_in, int& idx) {
  LayerP p;
  p.Wk = (const float*)d_in[idx++];  p.bk = (const float*)d_in[idx++];
  p.Wv = (const float*)d_in[idx++];  p.bv = (const float*)d_in[idx++];
  p.Wo = (const float*)d_in[idx++];  p.bo = (const float*)d_in[idx++];
  p.gammas = (const float*)d_in[idx++];
  p.ln1g = (const float*)d_in[idx++]; p.ln1b = (const float*)d_in[idx++];
  p.W1 = (const float*)d_in[idx++];  p.b1 = (const float*)d_in[idx++];
  p.W2 = (const float*)d_in[idx++];  p.b2 = (const float*)d_in[idx++];
  p.ln2g = (const float*)d_in[idx++]; p.ln2b = (const float*)d_in[idx++];
  return p;
}

extern "C" void kernel_launch(void* const* d_in, const int* in_sizes, int n_in,
                              void* d_out, int out_size, void* d_ws, size_t ws_size,
                              hipStream_t stream) {
  (void)in_sizes; (void)n_in; (void)out_size; (void)ws_size;
  const size_t nD = (size_t)AKT_BS * AKT_D;

  float* X  = (float*)d_ws;     // current x
  float* Y  = X  + nD;          // current y
  float* QB = Y  + nD;          // Q projection / attention output (aliased)
  float* KB = QB + nD;
  float* VB = KB + nD;
  float* T1 = VB + nD;          // Wo / W2 output
  float* X1 = T1 + nD;          // post-LN1 activation
  float* HF = X1 + nD;          // FFN hidden [BS, F]

  (void)hipMemcpyAsync(X, d_in[0], nD * sizeof(float), hipMemcpyDeviceToDevice, stream);
  (void)hipMemcpyAsync(Y, d_in[1], nD * sizeof(float), hipMemcpyDeviceToDevice, stream);

  int idx = 3;                  // skip question/interaction/difficulty embs
  LayerP pk[2], pq[4];
  for (int i = 0; i < 2; ++i) pk[i] = parse_layer(d_in, idx);
  for (int i = 0; i < 4; ++i) pq[i] = parse_layer(d_in, idx);

  auto gemm = [&](const float* A, const float* W, const float* bias, float* C,
                  int M, int N, int K, int relu) {
    dim3 grid(N / TN, M / TM);
    gemm_wmma_kernel<<<grid, 256, 0, stream>>>(A, W, bias, C, M, N, K, relu);
  };

  auto block = [&](const LayerP& L, const float* q, const float* k,
                   const float* v, float* dest, int apply_pos, int mask_flag) {
    gemm(q, L.Wk, L.bk, QB, AKT_BS, AKT_D, AKT_D, 0);
    gemm(k, L.Wk, L.bk, KB, AKT_BS, AKT_D, AKT_D, 0);
    gemm(v, L.Wv, L.bv, VB, AKT_BS, AKT_D, AKT_D, 0);
    dim3 ga(AKT_S / 16, AKT_B * AKT_H);
    akt_attn_kernel<<<ga, 32, 0, stream>>>(QB, KB, VB, L.gammas, QB, mask_flag);
    gemm(QB, L.Wo, L.bo, T1, AKT_BS, AKT_D, AKT_D, 0);
    residual_ln_kernel<<<AKT_BS, 32, 0, stream>>>(
        T1, q, L.ln1g, L.ln1b, apply_pos ? X1 : dest);
    if (apply_pos) {
      gemm(X1, L.W1, L.b1, HF, AKT_BS, AKT_F, AKT_D, 1);
      gemm(HF, L.W2, L.b2, T1, AKT_BS, AKT_D, AKT_F, 0);
      residual_ln_kernel<<<AKT_BS, 32, 0, stream>>>(T1, X1, L.ln2g, L.ln2b, dest);
    }
  };

  // encoder on y (params_k)
  block(pk[0], Y, Y, Y, Y, 1, 1);
  block(pk[1], Y, Y, Y, Y, 1, 1);
  // decoder on x (params_q): self (no FFN) / cross (FFN), alternating
  block(pq[0], X, X, X, X, 0, 1);
  block(pq[1], X, X, Y, X, 1, 0);
  block(pq[2], X, X, X, X, 0, 1);
  block(pq[3], X, X, Y, X, 1, 0);

  (void)hipMemcpyAsync(d_out, X, nD * sizeof(float), hipMemcpyDeviceToDevice, stream);
}
